// CausalSelfAttention_81372450390121
// MI455X (gfx1250) — matrix operand order repaired and checked by reading, not compiled, not run on hardware
//
#include <hip/hip_runtime.h>

typedef __attribute__((ext_vector_type(16))) __bf16 v16bf;
typedef __attribute__((ext_vector_type(8)))  __bf16 v8bf;
typedef __attribute__((ext_vector_type(8)))  float  v8f;

// ---------------------------------------------------------------- helpers

static __device__ __forceinline__ v8f wmma_bf16(v16bf a, v16bf b, v8f c) {
  // D = A(16x32 bf16) x B(32x16 bf16) + C(16x16 f32)
  return __builtin_amdgcn_wmma_f32_16x16x32_bf16(false, a, false, b,
                                                 (short)0, c, false, false);
}

static __device__ __forceinline__ v8f vzero8() {
  v8f z;
#pragma unroll
  for (int i = 0; i < 8; ++i) z[i] = 0.0f;
  return z;
}

// A fragment (16x32, bf16) from row-major LDS tile, stride ld elements.
// Layout: lane row = lane%16; lanes 0-15: K = 0..15,
// lanes 16-31: K = 16..31 (contiguous, matches B fragments).
static __device__ __forceinline__ v16bf load_a_frag_lds(const __bf16* p, int ld) {
  const int lane = threadIdx.x & 31;
  const int row = lane & 15, half = lane >> 4;
  const __bf16* r = p + row * ld;
  v8bf lo = *(const v8bf*)(r + half * 16);
  v8bf hi = *(const v8bf*)(r + 8 + half * 16);
  v16bf a;
#pragma unroll
  for (int i = 0; i < 8; ++i) { a[i] = lo[i]; a[i + 8] = hi[i]; }
  return a;
}

// ---------------------------------------------------------------- convert

__global__ void cvt_f32_to_bf16(const float* __restrict__ in,
                                __bf16* __restrict__ out, int n) {
  int i = blockIdx.x * blockDim.x + threadIdx.x;
  const int stride = gridDim.x * blockDim.x;
  for (; i < n; i += stride) out[i] = (__bf16)in[i];
}

// ---------------------------------------------------------------- QKV GEMM
// qkv[8192,3072] = Xb[8192,1024] @ Wa[1024,3072] + b_attn, scattered to
// Q/K/V [B=4,H=16,T=2048,D=64] bf16.
// Block tile 128x128, 8 waves (4m x 2n), wave tile 32x64 -> acc[2][4].

__global__ void __launch_bounds__(256) qkv_gemm_kernel(
    const __bf16* __restrict__ Xb, const __bf16* __restrict__ Wa,
    const float* __restrict__ b_attn,
    __bf16* __restrict__ Qb, __bf16* __restrict__ Kb, __bf16* __restrict__ Vb) {
  alignas(32) __shared__ __bf16 As[128 * 32];   // [m][k]
  alignas(32) __shared__ __bf16 Bs[128 * 32];   // [n][k] (transposed)

  const int tid = threadIdx.x;
  const int lane = tid & 31, w = tid >> 5;
  const int wm = w >> 1, wn = w & 1;
  const int half = lane >> 4, ln = lane & 15;
  const int mb = blockIdx.y * 128;
  const int nb = blockIdx.x * 128;

  v8f acc[2][4];
#pragma unroll
  for (int i = 0; i < 2; ++i)
#pragma unroll
    for (int j = 0; j < 4; ++j) acc[i][j] = vzero8();

  const int arow = tid >> 1, ak0 = (tid & 1) * 16;   // A: 16 elems/thread
  const int bkr = tid >> 3, bnc = (tid & 7) * 16;    // B: 16 elems/thread

  for (int kb = 0; kb < 1024; kb += 32) {
    __syncthreads();
    {
      const v8bf* src = (const v8bf*)(Xb + (size_t)(mb + arow) * 1024 + kb + ak0);
      *(v8bf*)(As + arow * 32 + ak0) = src[0];
      *(v8bf*)(As + arow * 32 + ak0 + 8) = src[1];
      const v8bf* wsrc = (const v8bf*)(Wa + (size_t)(kb + bkr) * 3072 + nb + bnc);
      v8bf w0 = wsrc[0], w1 = wsrc[1];
#pragma unroll
      for (int j = 0; j < 8; ++j) {
        Bs[(bnc + j) * 32 + bkr] = w0[j];
        Bs[(bnc + 8 + j) * 32 + bkr] = w1[j];
      }
      if (kb + 32 < 1024) {  // gfx1250 global_prefetch of next k-tile
        __builtin_prefetch(Xb + (size_t)(mb + arow) * 1024 + kb + 32 + ak0, 0, 1);
        __builtin_prefetch(Wa + (size_t)(kb + 32 + bkr) * 3072 + nb + bnc, 0, 1);
      }
    }
    __syncthreads();

    v16bf a0 = load_a_frag_lds(As + (wm * 32) * 32, 32);
    v16bf a1 = load_a_frag_lds(As + (wm * 32 + 16) * 32, 32);
#pragma unroll
    for (int j = 0; j < 4; ++j) {
      // B fragment: N = ln, K = i + 16*half -> contiguous 32B in Bs[n][k]
      v16bf bf = *(const v16bf*)(Bs + (wn * 64 + j * 16 + ln) * 32 + 16 * half);
      acc[0][j] = wmma_bf16(a0, bf, acc[0][j]);
      acc[1][j] = wmma_bf16(a1, bf, acc[1][j]);
    }
  }

#pragma unroll
  for (int mi = 0; mi < 2; ++mi)
#pragma unroll
    for (int ni = 0; ni < 4; ++ni)
#pragma unroll
      for (int r = 0; r < 8; ++r) {
        const int m_g = mb + wm * 32 + mi * 16 + r + 8 * half;
        const int c = nb + wn * 64 + ni * 16 + ln;
        const float v = acc[mi][ni][r] + b_attn[c];
        const int sel = c >> 10, cc = c & 1023;
        const int h = cc >> 6, d = cc & 63;
        const int bb = m_g >> 11, t = m_g & 2047;
        __bf16* dst = (sel == 0) ? Qb : (sel == 1) ? Kb : Vb;
        dst[(((size_t)bb * 16 + h) * 2048 + t) * 64 + d] = (__bf16)v;
      }
}

// ---------------------------------------------------------------- attention
// Flash-attention: grid (T/128, B*H), 256 threads = 8 waves, 16 q-rows/wave.
// Each staged 32-key K/V chunk now feeds 128 query rows.

__global__ void __launch_bounds__(256) attn_kernel(
    const __bf16* __restrict__ Qw, const __bf16* __restrict__ Kw,
    const __bf16* __restrict__ Vw, __bf16* __restrict__ Yw) {
  alignas(32) __shared__ __bf16 Ks[32 * 64];      // [key][d]
  alignas(32) __shared__ __bf16 Vt[64 * 32];      // [d][key]
  alignas(32) __shared__ __bf16 Pb[8][16 * 32];   // per-wave P tile [q][key]

  const int tid = threadIdx.x, lane = tid & 31, w = tid >> 5;
  const int half = lane >> 4, ln = lane & 15;
  const int bh = blockIdx.y;
  const int qb = blockIdx.x * 128;
  const int qw = qb + w * 16;

  const size_t hd = (size_t)bh * 2048 * 64;
  const __bf16* Qg = Qw + hd;
  const __bf16* Kg = Kw + hd;
  const __bf16* Vg = Vw + hd;

  // Q fragments for this wave's 16 rows (pre-scaled by 1/sqrt(D) = 0.125)
  v16bf qf[2];
  {
    const int qrow = qw + ln;
#pragma unroll
    for (int c = 0; c < 2; ++c) {
      const __bf16* r = Qg + (size_t)qrow * 64 + c * 32;
      v8bf lo = *(const v8bf*)(r + half * 16);
      v8bf hi = *(const v8bf*)(r + 8 + half * 16);
#pragma unroll
      for (int i = 0; i < 8; ++i) {
        qf[c][i]     = (__bf16)((float)lo[i] * 0.125f);
        qf[c][i + 8] = (__bf16)((float)hi[i] * 0.125f);
      }
    }
  }

  float mrow[8], lrow[8];
#pragma unroll
  for (int r = 0; r < 8; ++r) { mrow[r] = -1e30f; lrow[r] = 0.0f; }
  v8f o0 = vzero8(), o1 = vzero8(), o2 = vzero8(), o3 = vzero8();

  const int skey = tid >> 3;           // staging: key row 0..31
  const int sd0 = (tid & 7) * 8;       // staging: d base 0..56 step 8

  const int kend = qb + 128;
  for (int kb = 0; kb < kend; kb += 32) {
    __syncthreads();
    {
      v8bf kv = *(const v8bf*)(Kg + (size_t)(kb + skey) * 64 + sd0);
      *(v8bf*)(Ks + skey * 64 + sd0) = kv;
      v8bf vv = *(const v8bf*)(Vg + (size_t)(kb + skey) * 64 + sd0);
#pragma unroll
      for (int u = 0; u < 8; ++u) Vt[(sd0 + u) * 32 + skey] = vv[u];
      if (kb + 32 < kend) {
        __builtin_prefetch(Kg + (size_t)(kb + 32 + skey) * 64 + sd0, 0, 1);
        __builtin_prefetch(Vg + (size_t)(kb + 32 + skey) * 64 + sd0, 0, 1);
      }
    }
    __syncthreads();

    if (kb > qw + 15) continue;  // wave-uniform: chunk fully above diagonal

    // scores S[16q x 32key] = Q . K^T  (two 16x16 f32 accumulators)
    v8f s0 = vzero8(), s1 = vzero8();
#pragma unroll
    for (int c = 0; c < 2; ++c) {
      v16bf kf0 = *(const v16bf*)(Ks + ln * 64 + 32 * c + 16 * half);
      v16bf kf1 = *(const v16bf*)(Ks + (16 + ln) * 64 + 32 * c + 16 * half);
      s0 = wmma_bf16(qf[c], kf0, s0);
      s1 = wmma_bf16(qf[c], kf1, s1);
    }

    // causal mask + online softmax (rows r+8*half, keys = lane columns)
    float al[8];
#pragma unroll
    for (int r = 0; r < 8; ++r) {
      const int row = qw + r + 8 * half;
      const float a = (kb + ln <= row) ? s0[r] : -1e30f;
      const float b = (kb + 16 + ln <= row) ? s1[r] : -1e30f;
      float rm = fmaxf(a, b);
      rm = fmaxf(rm, __shfl_xor(rm, 1, 32));
      rm = fmaxf(rm, __shfl_xor(rm, 2, 32));
      rm = fmaxf(rm, __shfl_xor(rm, 4, 32));
      rm = fmaxf(rm, __shfl_xor(rm, 8, 32));
      const float mnew = fmaxf(mrow[r], rm);
      const float alpha = __expf(mrow[r] - mnew);
      const float p0 = __expf(a - mnew);
      const float p1 = __expf(b - mnew);
      s0[r] = p0; s1[r] = p1;
      float rs = p0 + p1;
      rs += __shfl_xor(rs, 1, 32);
      rs += __shfl_xor(rs, 2, 32);
      rs += __shfl_xor(rs, 4, 32);
      rs += __shfl_xor(rs, 8, 32);
      lrow[r] = lrow[r] * alpha + rs;
      mrow[r] = mnew;
      al[r] = alpha;
    }

    // rescale O, transpose P (C-layout -> A-layout) through wave-private LDS
    __bf16* pb = Pb[w];
#pragma unroll
    for (int r = 0; r < 8; ++r) {
      o0[r] *= al[r]; o1[r] *= al[r]; o2[r] *= al[r]; o3[r] *= al[r];
      pb[(r + 8 * half) * 32 + ln] = (__bf16)s0[r];
      pb[(r + 8 * half) * 32 + 16 + ln] = (__bf16)s1[r];
    }
    v16bf pf = load_a_frag_lds(pb, 32);

    // O += P . V  (V fragments contiguous over key in Vt[d][key])
    v16bf vf0 = *(const v16bf*)(Vt + (0 * 16 + ln) * 32 + 16 * half);
    v16bf vf1 = *(const v16bf*)(Vt + (1 * 16 + ln) * 32 + 16 * half);
    v16bf vf2 = *(const v16bf*)(Vt + (2 * 16 + ln) * 32 + 16 * half);
    v16bf vf3 = *(const v16bf*)(Vt + (3 * 16 + ln) * 32 + 16 * half);
    o0 = wmma_bf16(pf, vf0, o0);
    o1 = wmma_bf16(pf, vf1, o1);
    o2 = wmma_bf16(pf, vf2, o2);
    o3 = wmma_bf16(pf, vf3, o3);
  }

  // normalize + store y[B,T,C] as bf16
  const int bb = bh >> 4, h = bh & 15;
#pragma unroll
  for (int r = 0; r < 8; ++r) {
    const float inv = 1.0f / lrow[r];
    const int t = qw + r + 8 * half;
    const size_t base = ((size_t)bb * 2048 + t) * 1024 + h * 64;
    Yw[base + 0 * 16 + ln] = (__bf16)(o0[r] * inv);
    Yw[base + 1 * 16 + ln] = (__bf16)(o1[r] * inv);
    Yw[base + 2 * 16 + ln] = (__bf16)(o2[r] * inv);
    Yw[base + 3 * 16 + ln] = (__bf16)(o3[r] * inv);
  }
}

// ---------------------------------------------------------------- proj GEMM
// out[8192,1024] f32 = Yb[8192,1024] @ Wp[1024,1024] + b_proj
// Same 128x128 tiling as qkv GEMM.

__global__ void __launch_bounds__(256) proj_gemm_kernel(
    const __bf16* __restrict__ Yb, const __bf16* __restrict__ Wp,
    const float* __restrict__ b_proj, float* __restrict__ Out) {
  alignas(32) __shared__ __bf16 As[128 * 32];
  alignas(32) __shared__ __bf16 Bs[128 * 32];

  const int tid = threadIdx.x;
  const int lane = tid & 31, w = tid >> 5;
  const int wm = w >> 1, wn = w & 1;
  const int half = lane >> 4, ln = lane & 15;
  const int mb = blockIdx.y * 128;
  const int nb = blockIdx.x * 128;

  v8f acc[2][4];
#pragma unroll
  for (int i = 0; i < 2; ++i)
#pragma unroll
    for (int j = 0; j < 4; ++j) acc[i][j] = vzero8();

  const int arow = tid >> 1, ak0 = (tid & 1) * 16;
  const int bkr = tid >> 3, bnc = (tid & 7) * 16;

  for (int kb = 0; kb < 1024; kb += 32) {
    __syncthreads();
    {
      const v8bf* src = (const v8bf*)(Yb + (size_t)(mb + arow) * 1024 + kb + ak0);
      *(v8bf*)(As + arow * 32 + ak0) = src[0];
      *(v8bf*)(As + arow * 32 + ak0 + 8) = src[1];
      const v8bf* wsrc = (const v8bf*)(Wp + (size_t)(kb + bkr) * 1024 + nb + bnc);
      v8bf w0 = wsrc[0], w1 = wsrc[1];
#pragma unroll
      for (int j = 0; j < 8; ++j) {
        Bs[(bnc + j) * 32 + bkr] = w0[j];
        Bs[(bnc + 8 + j) * 32 + bkr] = w1[j];
      }
      if (kb + 32 < 1024) {
        __builtin_prefetch(Yb + (size_t)(mb + arow) * 1024 + kb + 32 + ak0, 0, 1);
        __builtin_prefetch(Wp + (size_t)(kb + 32 + bkr) * 1024 + nb + bnc, 0, 1);
      }
    }
    __syncthreads();

    v16bf a0 = load_a_frag_lds(As + (wm * 32) * 32, 32);
    v16bf a1 = load_a_frag_lds(As + (wm * 32 + 16) * 32, 32);
#pragma unroll
    for (int j = 0; j < 4; ++j) {
      v16bf bf = *(const v16bf*)(Bs + (wn * 64 + j * 16 + ln) * 32 + 16 * half);
      acc[0][j] = wmma_bf16(a0, bf, acc[0][j]);
      acc[1][j] = wmma_bf16(a1, bf, acc[1][j]);
    }
  }

#pragma unroll
  for (int mi = 0; mi < 2; ++mi)
#pragma unroll
    for (int ni = 0; ni < 4; ++ni)
#pragma unroll
      for (int r = 0; r < 8; ++r) {
        const int m_g = mb + wm * 32 + mi * 16 + r + 8 * half;
        const int c = nb + wn * 64 + ni * 16 + ln;
        Out[(size_t)m_g * 1024 + c] = acc[mi][ni][r] + b_proj[c];
      }
}

// ---------------------------------------------------------------- launcher

extern "C" void kernel_launch(void* const* d_in, const int* in_sizes, int n_in,
                              void* d_out, int out_size, void* d_ws, size_t ws_size,
                              hipStream_t stream) {
  (void)in_sizes; (void)n_in; (void)out_size; (void)ws_size;
  const float* x      = (const float*)d_in[0];
  const float* W_attn = (const float*)d_in[1];
  const float* b_attn = (const float*)d_in[2];
  const float* W_proj = (const float*)d_in[3];
  const float* b_proj = (const float*)d_in[4];
  float* out = (float*)d_out;

  char* ws = (char*)d_ws;
  __bf16* Xb = (__bf16*)(ws + 0);          // 8192*1024*2   = 16777216
  __bf16* Wa = (__bf16*)(ws + 16777216);   // 1024*3072*2   =  6291456
  __bf16* Wp = (__bf16*)(ws + 23068672);   // 1024*1024*2   =  2097152
  __bf16* Qb = (__bf16*)(ws + 25165824);   // 4*16*2048*64*2 = 16777216
  __bf16* Kb = (__bf16*)(ws + 41943040);
  __bf16* Vb = (__bf16*)(ws + 58720256);
  __bf16* Yb = (__bf16*)(ws + 75497472);   // total 92274688 bytes

  cvt_f32_to_bf16<<<2048, 256, 0, stream>>>(x, Xb, 8192 * 1024);
  cvt_f32_to_bf16<<<2048, 256, 0, stream>>>(W_attn, Wa, 1024 * 3072);
  cvt_f32_to_bf16<<<1024, 256, 0, stream>>>(W_proj, Wp, 1024 * 1024);

  qkv_gemm_kernel<<<dim3(24, 64), 256, 0, stream>>>(Xb, Wa, b_attn, Qb, Kb, Vb);
  attn_kernel<<<dim3(16, 64), 256, 0, stream>>>(Qb, Kb, Vb, Yb);
  proj_gemm_kernel<<<dim3(8, 64), 256, 0, stream>>>(Yb, Wp, b_proj, out);
}
